// CrossUpdate_47536698032921
// MI455X (gfx1250) — compile-verified, matrix operand order checked by reading, not run on hardware
//
#include <hip/hip_runtime.h>

// ---------------------------------------------------------------------------
// CDNA5 (gfx1250) CrossUpdate (DCN-GRU tracker update).
// All conv / DCN contractions use v_wmma_f32_16x16x32_bf16 (wave32, 16x16
// tiles, K-step 32, fp32 accumulate). Reduction order kk = tap*CIN + ci so
// each 32-wide K step has a constant tap and a contiguous channel block.
// Weights are repacked per conv into bf16 [o][kk] so B fragments are two
// aligned 16B vector loads. Each wave computes 32 pixels x (16*NTILES)
// couts so the (expensive) A gather/bilinear work is amortized over NTILES
// B tiles.
// ---------------------------------------------------------------------------

typedef __attribute__((ext_vector_type(16))) __bf16 v16bf;
typedef __attribute__((ext_vector_type(8)))  __bf16 v8bf;
typedef __attribute__((ext_vector_type(8)))  float  v8f;

constexpr int Hc    = 96;
constexpr int Wc    = 96;
constexpr int Nb    = 2;
constexpr int HWc   = Hc * Wc;          // 9216 (multiple of 32)
constexpr int MtotC = Nb * HWc;         // 18432 = 576 tiles of 32

__device__ __forceinline__ float sigmoid_f(float v) {
    return 1.0f / (1.0f + __expf(-v));
}

// ---------------------------------------------------------------------------
// Weight repack: OIHW fp32 -> bf16 [o][kk], kk = r*CIN + ci, o padded to 32.
// ---------------------------------------------------------------------------
template<int CIN, int COUT>
__global__ __launch_bounds__(256)
void repack_w_kernel(const float* __restrict__ w, __bf16* __restrict__ wpk)
{
    constexpr int K9    = CIN * 9;
    constexpr int COUTP = (COUT + 31) & ~31;
    const int idx = blockIdx.x * 256 + threadIdx.x;
    if (idx >= COUTP * K9) return;
    const int o  = idx / K9;
    const int kk = idx - o * K9;
    const int r  = kk / CIN;
    const int ci = kk - r * CIN;
    const float v = (o < COUT) ? w[((size_t)o * CIN + ci) * 9 + r] : 0.0f;
    wpk[idx] = (__bf16)v;
}

// ---------------------------------------------------------------------------
// Implicit-GEMM 3x3 SAME conv. Input is a virtual channel-concat
// (in0: CIN0 ch, in1: CIN-CIN0 ch). One wave -> 32 pixels x 16*NTILES couts;
// A fragments are built once per K-step and reused for all NTILES WMMAs.
// ACT: 0 none, 1 sigmoid, 2 tanh.
// ---------------------------------------------------------------------------
template<int CIN, int CIN0, int COUT, int ACT, int NTILES>
__global__ __launch_bounds__(32)
void conv3x3_wmma_kernel(const float* __restrict__ in0,
                         const float* __restrict__ in1,
                         const __bf16* __restrict__ wpk,
                         const float* __restrict__ b,
                         float* __restrict__ out)
{
    constexpr int K9 = CIN * 9;        // multiple of 32 (CIN % 32 == 0)
    const int tileM = blockIdx.x * 32;
    const int tileN = blockIdx.y * (16 * NTILES);
    const int lane  = threadIdx.x;     // wave32
    const int half  = lane >> 4;
    const int lp    = lane & 15;

    // pixel coords of the two A rows this lane feeds (M = lp per half)
    int pn[2], py[2], px[2];
#pragma unroll
    for (int mi = 0; mi < 2; ++mi) {
        const int m = tileM + mi * 16 + lp;
        pn[mi] = m / HWc;
        const int rem = m - pn[mi] * HWc;
        py[mi] = rem / Wc;
        px[mi] = rem - py[mi] * Wc;
    }
    int on[NTILES];
    const __bf16* bptr[NTILES];
#pragma unroll
    for (int ni = 0; ni < NTILES; ++ni) {
        on[ni]   = tileN + ni * 16 + lp;
        bptr[ni] = wpk + (size_t)on[ni] * K9 + half * 8;
    }

    v8f acc[2][NTILES] = {};

#pragma unroll 1
    for (int r = 0; r < 9; ++r) {
        const int dy = r / 3 - 1;
        const int dx = r % 3 - 1;

        // per-row tap address + zero-pad mask (hoisted out of element loop)
        int   aoff[2];
        float amask[2];
#pragma unroll
        for (int mi = 0; mi < 2; ++mi) {
            const int iy = py[mi] + dy;
            const int ix = px[mi] + dx;
            const bool inb = ((unsigned)iy < (unsigned)Hc) &&
                             ((unsigned)ix < (unsigned)Wc);
            amask[mi] = inb ? 1.0f : 0.0f;
            const int iyc = min(max(iy, 0), Hc - 1);
            const int ixc = min(max(ix, 0), Wc - 1);
            aoff[mi] = iyc * Wc + ixc;
        }

#pragma unroll 1
        for (int cb = 0; cb < CIN; cb += 32) {
            const int k0 = r * CIN + cb;

            // ---- A fragments: 16 strided loads per row, branch-free ------
            v16bf af[2];
#pragma unroll
            for (int mi = 0; mi < 2; ++mi) {
                const float* sb = (cb < CIN0)
                    ? in0 + ((size_t)pn[mi] * CIN0 + cb) * HWc
                    : in1 + ((size_t)pn[mi] * (CIN - CIN0) + (cb - CIN0)) * HWc;
                const float* sp = sb + aoff[mi];
                const float  msk = amask[mi];
#pragma unroll
                for (int e = 0; e < 16; ++e) {
                    const int coff = (e >> 3) * 16 + half * 8 + (e & 7);
                    af[mi][e] = (__bf16)(sp[(size_t)coff * HWc] * msk);
                }
            }
            if (k0 + 32 < K9)
                __builtin_prefetch(bptr[0] + k0 + 32, 0, 1);

            // ---- B fragments + WMMAs: A reused across NTILES cout tiles --
#pragma unroll
            for (int ni = 0; ni < NTILES; ++ni) {
                const v8bf* p = (const v8bf*)(bptr[ni] + k0);
                const v8bf lo = p[0];   // kk = k0 + half*8 + 0..7
                const v8bf hi = p[2];   // kk = k0 + 16 + half*8 + 0..7
                const v16bf bfv = __builtin_shufflevector(lo, hi,
                    0,1,2,3,4,5,6,7,8,9,10,11,12,13,14,15);
#pragma unroll
                for (int mi = 0; mi < 2; ++mi)
                    acc[mi][ni] = __builtin_amdgcn_wmma_f32_16x16x32_bf16(
                        false, af[mi], false, bfv,
                        (short)0, acc[mi][ni], false, false);
            }
        }
    }

    // ---- Epilogue: bias + activation; 8 contiguous stores per fragment ---
#pragma unroll
    for (int mi = 0; mi < 2; ++mi) {
        const int mb   = tileM + mi * 16 + half * 8;   // 8-aligned row base
        const int nimg = mb / HWc;                     // constant across 8 rows
        const int rem  = mb - nimg * HWc;
#pragma unroll
        for (int ni = 0; ni < NTILES; ++ni) {
            const int o = on[ni];
            if ((COUT & 31) != 0 && o >= COUT) continue;
            const float bias = b[o];
            float* op = out + ((size_t)nimg * COUT + o) * HWc + rem;
#pragma unroll
            for (int v = 0; v < 8; ++v) {
                float val = acc[mi][ni][v] + bias;
                if (ACT == 1)      val = sigmoid_f(val);
                else if (ACT == 2) val = tanhf(val);
                op[v] = val;
            }
        }
    }
}

// ---------------------------------------------------------------------------
// DCN main contraction, COUT = 64 done by ONE wave (NTILES=4) so the
// expensive deformable bilinear A sampling is computed exactly once.
// Per 32-wide K step the tap j is constant: the bilinear setup (4 clamped
// offsets + 4 mask*lerp weights, premultiplied by sigmoid(mask)) is computed
// once per output row per tap; the element loop is 4 strided loads + 4 FMA.
// Per-pixel tap state staged in LDS once per tile (reused CIN times).
// ---------------------------------------------------------------------------
template<int CIN, int CIN0, int ACT>
__global__ __launch_bounds__(32)
void dcn_wmma_kernel(const float* __restrict__ in0,
                     const float* __restrict__ in1,
                     const float* __restrict__ om,
                     const __bf16* __restrict__ wpk,
                     const float* __restrict__ b,
                     float* __restrict__ out)
{
    constexpr int K9     = CIN * 9;
    constexpr int COUT   = 64;
    constexpr int NTILES = 4;
    const int tileM = blockIdx.x * 32;
    const int lane  = threadIdx.x;
    const int half  = lane >> 4;
    const int lp    = lane & 15;

    __shared__ float s_py[32 * 9];
    __shared__ float s_px[32 * 9];
    __shared__ float s_mk[32 * 9];
    __shared__ int   s_n[32];

    {   // one lane per tile pixel: stage 27 offset-map values
        const int m   = tileM + lane;
        const int n   = m / HWc;
        const int rem = m - n * HWc;
        const int oy  = rem / Wc;
        const int ox  = rem - oy * Wc;
        s_n[lane] = n;
        const float* omp = om + (size_t)n * 27 * HWc + rem;
#pragma unroll
        for (int j = 0; j < 9; ++j) {
            const float dyv = omp[(size_t)j * HWc];
            const float dxv = omp[(size_t)(9 + j) * HWc];
            const float mkv = omp[(size_t)(18 + j) * HWc];
            s_py[lane * 9 + j] = (float)oy + (float)(j / 3) - 1.0f + dyv;
            s_px[lane * 9 + j] = (float)ox + (float)(j % 3) - 1.0f + dxv;
            s_mk[lane * 9 + j] = sigmoid_f(mkv);
        }
    }
    __syncthreads();

    int on[NTILES];
    const __bf16* bptr[NTILES];
#pragma unroll
    for (int ni = 0; ni < NTILES; ++ni) {
        on[ni]   = ni * 16 + lp;
        bptr[ni] = wpk + (size_t)on[ni] * K9 + half * 8;
    }

    v8f acc[2][NTILES] = {};

#pragma unroll 1
    for (int j = 0; j < 9; ++j) {
        // per-row bilinear setup (constant across the channel block loop)
        int   o00[2], o01[2], o10[2], o11[2];
        float w00[2], w01[2], w10[2], w11[2];
        int   nimg[2];
#pragma unroll
        for (int mi = 0; mi < 2; ++mi) {
            const int pidx = mi * 16 + lp;
            nimg[mi] = s_n[pidx];
            const float fy = s_py[pidx * 9 + j];
            const float fx = s_px[pidx * 9 + j];
            const float mk = s_mk[pidx * 9 + j];
            const float y0f = floorf(fy), x0f = floorf(fx);
            const int   y0  = (int)y0f,  x0  = (int)x0f;
            const float wy  = fy - y0f,  wx  = fx - x0f;
            const bool by0 = (unsigned)y0       < (unsigned)Hc;
            const bool by1 = (unsigned)(y0 + 1) < (unsigned)Hc;
            const bool bx0 = (unsigned)x0       < (unsigned)Wc;
            const bool bx1 = (unsigned)(x0 + 1) < (unsigned)Wc;
            const int y0c = min(max(y0, 0),     Hc - 1);
            const int y1c = min(max(y0 + 1, 0), Hc - 1);
            const int x0c = min(max(x0, 0),     Wc - 1);
            const int x1c = min(max(x0 + 1, 0), Wc - 1);
            o00[mi] = y0c * Wc + x0c;  o01[mi] = y0c * Wc + x1c;
            o10[mi] = y1c * Wc + x0c;  o11[mi] = y1c * Wc + x1c;
            w00[mi] = (by0 && bx0) ? (1.0f - wy) * (1.0f - wx) * mk : 0.0f;
            w01[mi] = (by0 && bx1) ? (1.0f - wy) * wx          * mk : 0.0f;
            w10[mi] = (by1 && bx0) ? wy * (1.0f - wx)          * mk : 0.0f;
            w11[mi] = (by1 && bx1) ? wy * wx                   * mk : 0.0f;
        }

#pragma unroll 1
        for (int cb = 0; cb < CIN; cb += 32) {
            const int k0 = j * CIN + cb;

            v16bf af[2];
#pragma unroll
            for (int mi = 0; mi < 2; ++mi) {
                const float* sb = (cb < CIN0)
                    ? in0 + ((size_t)nimg[mi] * CIN0 + cb) * HWc
                    : in1 + ((size_t)nimg[mi] * (CIN - CIN0) + (cb - CIN0)) * HWc;
#pragma unroll
                for (int e = 0; e < 16; ++e) {
                    const int coff = (e >> 3) * 16 + half * 8 + (e & 7);
                    const float* sp = sb + (size_t)coff * HWc;
                    const float v = sp[o00[mi]] * w00[mi] + sp[o01[mi]] * w01[mi]
                                  + sp[o10[mi]] * w10[mi] + sp[o11[mi]] * w11[mi];
                    af[mi][e] = (__bf16)v;
                }
            }
            if (k0 + 32 < K9)
                __builtin_prefetch(bptr[0] + k0 + 32, 0, 1);

#pragma unroll
            for (int ni = 0; ni < NTILES; ++ni) {
                const v8bf* p = (const v8bf*)(bptr[ni] + k0);
                const v8bf lo = p[0];
                const v8bf hi = p[2];
                const v16bf bfv = __builtin_shufflevector(lo, hi,
                    0,1,2,3,4,5,6,7,8,9,10,11,12,13,14,15);
#pragma unroll
                for (int mi = 0; mi < 2; ++mi)
                    acc[mi][ni] = __builtin_amdgcn_wmma_f32_16x16x32_bf16(
                        false, af[mi], false, bfv,
                        (short)0, acc[mi][ni], false, false);
            }
        }
    }

#pragma unroll
    for (int mi = 0; mi < 2; ++mi) {
        const int mb  = tileM + mi * 16 + half * 8;
        const int nI  = mb / HWc;
        const int rem = mb - nI * HWc;
#pragma unroll
        for (int ni = 0; ni < NTILES; ++ni) {
            const int o = on[ni];
            const float bias = b[o];
            float* op = out + ((size_t)nI * COUT + o) * HWc + rem;
#pragma unroll
            for (int v = 0; v < 8; ++v) {
                float val = acc[mi][ni][v] + bias;
                if (ACT == 1)      val = sigmoid_f(val);
                else if (ACT == 2) val = tanhf(val);
                op[v] = val;
            }
        }
    }
}

// ---------------------------------------------------------------------------
// Neighborhood correlation + softmax + attention sample (7x7 taps, dil 2,
// C = 64). Per-pixel-varying B matrix -> not a GEMM. One thread per pixel;
// the f2 window reloads are L1/L2 resident.
// ---------------------------------------------------------------------------
__global__ __launch_bounds__(256)
void ncf_kernel(const float* __restrict__ f1,
                const float* __restrict__ f2,
                float* __restrict__ out)
{
    constexpr int C = 64;
    const int p = blockIdx.x * blockDim.x + threadIdx.x;
    if (p >= MtotC) return;
    const int n   = p / HWc;
    const int rem = p - n * HWc;
    const int y   = rem / Wc;
    const int x   = rem - y * Wc;

    const float* f1p = f1 + (size_t)n * C * HWc + rem;
    const float* f2b = f2 + (size_t)n * C * HWc;

    float corr[49];
    float cmax = -1e30f;
    int k = 0;
#pragma unroll 1
    for (int i = 0; i < 7; ++i) {
        const int yy = y + i * 2 - 6;
#pragma unroll 1
        for (int jx = 0; jx < 7; ++jx, ++k) {
            const int xx = x + jx * 2 - 6;
            float s = 0.0f;
            if ((unsigned)yy < (unsigned)Hc && (unsigned)xx < (unsigned)Wc) {
                const float* q = f2b + yy * Wc + xx;
#pragma unroll 4
                for (int c = 0; c < C; ++c)
                    s += f1p[(size_t)c * HWc] * q[(size_t)c * HWc];
            }
            corr[k] = s * 0.125f;          // 1/sqrt(64)
            cmax = fmaxf(cmax, corr[k]);
        }
    }
    float ssum = 0.0f;
#pragma unroll
    for (k = 0; k < 49; ++k) {
        corr[k] = __expf(corr[k] - cmax);
        ssum += corr[k];
    }
    const float inv = 1.0f / ssum;

#pragma unroll 1
    for (int c = 0; c < C; ++c) {
        const float* qc = f2b + (size_t)c * HWc;
        float accv = 0.0f;
        k = 0;
#pragma unroll 1
        for (int i = 0; i < 7; ++i) {
            const int yy = y + i * 2 - 6;
#pragma unroll
            for (int jx = 0; jx < 7; ++jx, ++k) {
                const int xx = x + jx * 2 - 6;
                const float v = ((unsigned)yy < (unsigned)Hc &&
                                 (unsigned)xx < (unsigned)Wc)
                                    ? qc[yy * Wc + xx] : 0.0f;
                accv += corr[k] * v;
            }
        }
        out[(size_t)n * C * HWc + (size_t)c * HWc + rem] = accv * inv;
    }
}

// out = (1-z)*t + z*xt  (z sigmoid'd and xt tanh'd in the GEMM epilogues)
__global__ __launch_bounds__(256)
void gru_combine_kernel(const float* __restrict__ z,
                        const float* __restrict__ t,
                        const float* __restrict__ xt,
                        float* __restrict__ out, int total)
{
    const int i = blockIdx.x * blockDim.x + threadIdx.x;
    if (i >= total) return;
    const float zv = z[i];
    out[i] = (1.0f - zv) * t[i] + zv * xt[i];
}

// ---------------------------------------------------------------------------
// Orchestration
// ---------------------------------------------------------------------------
extern "C" void kernel_launch(void* const* d_in, const int* in_sizes, int n_in,
                              void* d_out, int out_size, void* d_ws, size_t ws_size,
                              hipStream_t stream)
{
    (void)in_sizes; (void)n_in; (void)out_size; (void)ws_size;

    const float* x        = (const float*)d_in[0];
    const float* tmpl     = (const float*)d_in[1];
    const float* w_in     = (const float*)d_in[2];
    const float* b_in     = (const float*)d_in[3];
    const float* w_out    = (const float*)d_in[4];
    const float* b_out    = (const float*)d_in[5];
    const float* enh_z_w  = (const float*)d_in[6];
    const float* enh_z_b  = (const float*)d_in[7];
    const float* enh_z_ow = (const float*)d_in[8];
    const float* enh_z_ob = (const float*)d_in[9];
    const float* enh_h_w  = (const float*)d_in[10];
    const float* enh_h_b  = (const float*)d_in[11];
    const float* enh_h_ow = (const float*)d_in[12];
    const float* enh_h_ob = (const float*)d_in[13];
    const float* upd_z_w  = (const float*)d_in[14];
    const float* upd_z_b  = (const float*)d_in[15];
    const float* upd_z_ow = (const float*)d_in[16];
    const float* upd_z_ob = (const float*)d_in[17];
    const float* upd_h_w  = (const float*)d_in[18];
    const float* upd_h_b  = (const float*)d_in[19];
    const float* upd_h_ow = (const float*)d_in[20];
    const float* upd_h_ob = (const float*)d_in[21];

    constexpr int CIN = 256, CH = 64;
    constexpr size_t PLANE = (size_t)Nb * HWc;

    // workspace layout (16B-aligned sections)
    float* ws   = (float*)d_ws;
    float* xh   = ws;                            // [Nb,CH,H,W]
    float* om   = xh   + PLANE * CH;             // [Nb,27,H,W]
    float* zb   = om   + PLANE * 27;
    float* rb   = zb   + PLANE * CH;
    float* xtb  = rb   + PLANE * CH;
    float* xenh = xtb  + PLANE * CH;
    __bf16* wpk = (__bf16*)(xenh + PLANE * CH);  // repacked weights (<=300KB)

    float* outp      = (float*)d_out;                 // [Nb,CIN,H,W]
    float* new_templ = outp + PLANE * CIN;            // [Nb,CH,H,W]

    const dim3 wblk(32);
    const dim3 g_m1  (MtotC / 32, 1);      // whole cout range in one wave
    const dim3 g_m4  (MtotC / 32, 4);      // conv_out: 4 x 64 couts
    const dim3 eblk(256);
    const dim3 g_pix((MtotC + 255) / 256);
    const dim3 g_elem((int)((PLANE * CH + 255) / 256));

    auto rpgrid = [](int coutp, int k9) { return dim3((coutp * k9 + 255) / 256); };

    // 1) xh = conv(x, w_in) + b_in          (COUT=64 in one wave, NTILES=4)
    repack_w_kernel<256, 64><<<rpgrid(64, 2304), eblk, 0, stream>>>(w_in, wpk);
    conv3x3_wmma_kernel<256, 256, 64, 0, 4><<<g_m1, wblk, 0, stream>>>(
        x, x, wpk, b_in, xh);

    // ---------------- stargru #1: x_enh = stargru(template, xh) -----------
    repack_w_kernel<128, 27><<<rpgrid(32, 1152), eblk, 0, stream>>>(enh_z_ow, wpk);
    conv3x3_wmma_kernel<128, 64, 27, 0, 2><<<g_m1, wblk, 0, stream>>>(
        tmpl, xh, wpk, enh_z_ob, om);
    repack_w_kernel<128, 64><<<rpgrid(64, 1152), eblk, 0, stream>>>(enh_z_w, wpk);
    dcn_wmma_kernel<128, 64, 1><<<g_m1, wblk, 0, stream>>>(
        tmpl, xh, om, wpk, enh_z_b, zb);
    ncf_kernel<<<g_pix, eblk, 0, stream>>>(xh, tmpl, rb);
    repack_w_kernel<128, 27><<<rpgrid(32, 1152), eblk, 0, stream>>>(enh_h_ow, wpk);
    conv3x3_wmma_kernel<128, 64, 27, 0, 2><<<g_m1, wblk, 0, stream>>>(
        rb, xh, wpk, enh_h_ob, om);
    repack_w_kernel<128, 64><<<rpgrid(64, 1152), eblk, 0, stream>>>(enh_h_w, wpk);
    dcn_wmma_kernel<128, 64, 2><<<g_m1, wblk, 0, stream>>>(
        rb, xh, om, wpk, enh_h_b, xtb);
    gru_combine_kernel<<<g_elem, eblk, 0, stream>>>(
        zb, xh, xtb, xenh, (int)(PLANE * CH));

    // ---------------- stargru #2: new_template = stargru(xh, template) ----
    repack_w_kernel<128, 27><<<rpgrid(32, 1152), eblk, 0, stream>>>(upd_z_ow, wpk);
    conv3x3_wmma_kernel<128, 64, 27, 0, 2><<<g_m1, wblk, 0, stream>>>(
        xh, tmpl, wpk, upd_z_ob, om);
    repack_w_kernel<128, 64><<<rpgrid(64, 1152), eblk, 0, stream>>>(upd_z_w, wpk);
    dcn_wmma_kernel<128, 64, 1><<<g_m1, wblk, 0, stream>>>(
        xh, tmpl, om, wpk, upd_z_b, zb);
    ncf_kernel<<<g_pix, eblk, 0, stream>>>(tmpl, xh, rb);
    repack_w_kernel<128, 27><<<rpgrid(32, 1152), eblk, 0, stream>>>(upd_h_ow, wpk);
    conv3x3_wmma_kernel<128, 64, 27, 0, 2><<<g_m1, wblk, 0, stream>>>(
        rb, tmpl, wpk, upd_h_ob, om);
    repack_w_kernel<128, 64><<<rpgrid(64, 1152), eblk, 0, stream>>>(upd_h_w, wpk);
    dcn_wmma_kernel<128, 64, 2><<<g_m1, wblk, 0, stream>>>(
        rb, tmpl, om, wpk, upd_h_b, xtb);
    gru_combine_kernel<<<g_elem, eblk, 0, stream>>>(
        zb, tmpl, xtb, new_templ, (int)(PLANE * CH));

    // 4) out = conv(x_enh, w_out) + b_out   (256 couts: 4 blocks x NTILES=4)
    repack_w_kernel<64, 256><<<rpgrid(256, 576), eblk, 0, stream>>>(w_out, wpk);
    conv3x3_wmma_kernel<64, 64, 256, 0, 4><<<g_m4, wblk, 0, stream>>>(
        xenh, xenh, wpk, b_out, outp);
}
